// RadarGRU_68942815036007
// MI455X (gfx1250) — matrix operand order repaired and checked by reading, not compiled, not run on hardware
//
#include <hip/hip_runtime.h>
#include <hip/hip_bf16.h>

#define H 1024
#define G 3072            // 3*H
#define B_ 64
#define T_ 512
#define BT (B_ * T_)      // 32768
#define NT_W 192          // N tiles in weight space (3072/16)
#define NB_SCAN 16        // blocks in the persistent scan kernel

typedef __attribute__((ext_vector_type(16))) __bf16 v16bf;
typedef __attribute__((ext_vector_type(8)))  float  v8f;

struct F16x16 { uint4 lo, hi; };   // 32 bytes == one v16bf

#if defined(__AMDGCN__) && __has_builtin(__builtin_amdgcn_global_load_async_to_lds_b128) && \
    __has_builtin(__builtin_amdgcn_s_wait_asynccnt)
#define USE_ASYNC_LDS 1
typedef int i4v __attribute__((vector_size(16)));           // GCC-style int4 (builtin pointee type)
typedef __attribute__((address_space(1))) i4v* as1_i4;      // global (prints as "__device__")
typedef __attribute__((address_space(3))) i4v* as3_i4;      // LDS
#else
#define USE_ASYNC_LDS 0
#endif

__device__ __forceinline__ unsigned short f2bf(float f) {
  unsigned u = __builtin_bit_cast(unsigned, f);
  u += 0x7fffu + ((u >> 16) & 1u);           // round-to-nearest-even
  return (unsigned short)(u >> 16);
}

// A-fragment: lane holds row m = lane%16; K = {k0..k0+7} U {k0+16..k0+23}, k0 = 8*(lane/16)
__device__ __forceinline__ v16bf ld_afrag(const unsigned short* p) {
  F16x16 t;
  t.lo = *(const uint4*)p;
  t.hi = *(const uint4*)(p + 16);
  return __builtin_bit_cast(v16bf, t);
}

// 16-byte global->LDS copy: async DMA path on CDNA5 when available
__device__ __forceinline__ void cp16(unsigned short* l, const unsigned short* g) {
#if USE_ASYNC_LDS
  __builtin_amdgcn_global_load_async_to_lds_b128((as1_i4)g, (as3_i4)l, 0, 0);
#else
  *(uint4*)l = *(const uint4*)g;
#endif
}

__device__ __forceinline__ void wait_cp() {
#if USE_ASYNC_LDS
  __builtin_amdgcn_s_wait_asynccnt(0);
#endif
  __syncthreads();
}

__device__ __forceinline__ float fsigmoid(float x) { return 1.0f / (1.0f + __expf(-x)); }
__device__ __forceinline__ float ftanh_(float x) {
  float e = __expf(-2.0f * x);
  return (1.0f - e) / (1.0f + e);
}

// ---------------- conversion / swizzle kernels ----------------

__global__ void k_convert_x(const float* __restrict__ x, unsigned short* __restrict__ xbf) {
  size_t idx = (size_t)blockIdx.x * blockDim.x + threadIdx.x;
  if (idx >= (size_t)BT * 128) return;
  int d = (int)(idx & 127);
  size_t bt = idx >> 7;
  float v = (d < 118) ? x[bt * 118 + d] : 0.0f;
  xbf[idx] = f2bf(v);
}

// W (3072 x Kreal, row-major) -> per-(kt,nt) 1KB B-fragment tiles in documented
// B 32x16 VGPR order: dword v of lane holds K = 32*kt + 16*(v/4) + 8*(lane/16) + 2*(v%4) (+1), N = nt*16+lane%16.
__global__ void k_swizzle_w(const float* __restrict__ W, int Kreal, int KT,
                            unsigned* __restrict__ out) {
  int idx = blockIdx.x * blockDim.x + threadIdx.x;
  int total = KT * NT_W * 32;
  if (idx >= total) return;
  int lane = idx & 31, tile = idx >> 5;
  int nt = tile % NT_W, kt = tile / NT_W;
  int half = lane >> 4;
  int n = nt * 16 + (lane & 15);
  unsigned* dst = out + (size_t)idx * 8;
#pragma unroll
  for (int v = 0; v < 8; ++v) {
    int k0 = kt * 32 + (v >> 2) * 16 + half * 8 + 2 * (v & 3);
    float f0 = (k0     < Kreal) ? W[(size_t)n * Kreal + k0]     : 0.0f;
    float f1 = (k0 + 1 < Kreal) ? W[(size_t)n * Kreal + k0 + 1] : 0.0f;
    dst[v] = (unsigned)f2bf(f0) | ((unsigned)f2bf(f1) << 16);
  }
}

// ---------------- batched input projection GEMM (LDS double-buffered) ----------------
// xg[bt,g] = sum_k A_bf[bt,k]*W[g,k] + b_ih[g].  Block tile 64(M) x 128(N).
__global__ void __launch_bounds__(256) k_input_gemm(
    const unsigned short* __restrict__ A, int Kpad,
    const unsigned short* __restrict__ w_sw,
    const float* __restrict__ bih, float* __restrict__ xg) {
  __shared__ alignas(32) unsigned short lA[2][64 * 32];   // 4KB x2
  __shared__ alignas(32) unsigned short lB[2][8 * 512];   // 8KB x2

  const int tid = threadIdx.x;
  const int w = tid >> 5, lane = tid & 31, lane16 = lane & 15, half = lane >> 4;
  const int m16 = (w & 3) * 16, sel = w >> 2;
  const int mbase = blockIdx.x * 64;
  const int ntb0 = blockIdx.y * 8;
  const int KT = Kpad / 32;

  // staging assignment: A -> thread copies 16B of row tid/4; B -> wave w copies its subtile
  const int ar = tid >> 2, ac = (tid & 3) * 8;
  const unsigned short* gA = A + (size_t)(mbase + ar) * Kpad + ac;
  const unsigned short* gB = w_sw + ((size_t)(ntb0 + w) * 32 + lane) * 16;
  const size_t BSTRIDE = (size_t)NT_W * 512;   // elements per kt step

  v8f acc[4] = {};

  // prologue: stage kt=0 into buffer 0
  cp16(&lA[0][ar * 32 + ac], gA);
  cp16(&lB[0][(w * 32 + lane) * 16], gB);
  cp16(&lB[0][(w * 32 + lane) * 16 + 8], gB + 8);
  wait_cp();

  for (int kt = 0; kt < KT; ++kt) {
    if (kt + 1 < KT) {
      const int nb2 = (kt + 1) & 1;
      const unsigned short* ga = gA + (size_t)(kt + 1) * 32;
      const unsigned short* gb = gB + (size_t)(kt + 1) * BSTRIDE;
      cp16(&lA[nb2][ar * 32 + ac], ga);
      cp16(&lB[nb2][(w * 32 + lane) * 16], gb);
      cp16(&lB[nb2][(w * 32 + lane) * 16 + 8], gb + 8);
    }
    const int cb = kt & 1;
    v16bf a = ld_afrag(&lA[cb][(m16 + lane16) * 32 + half * 8]);
    const unsigned short* b_l = &lB[cb][(sel * 4 * 32 + lane) * 16];
#pragma unroll
    for (int j = 0; j < 4; ++j) {
      v16bf b = *(const v16bf*)(b_l + j * 512);
      acc[j] = __builtin_amdgcn_wmma_f32_16x16x32_bf16(false, a, false, b,
                                                       (short)0, acc[j], false, false);
    }
    wait_cp();   // staged kt+1 visible; all waves done reading buffer cb
  }

#pragma unroll
  for (int j = 0; j < 4; ++j) {
    const int col = (ntb0 + sel * 4 + j) * 16 + lane16;
    const float bv = bih[col];
#pragma unroll
    for (int r = 0; r < 8; ++r) {
      const int row = mbase + m16 + r + half * 8;
      xg[(size_t)row * G + col] = acc[j][r] + bv;
    }
  }
}

// ---------------- persistent recurrent scan ----------------

__device__ __forceinline__ void grid_bar(unsigned* ctr, unsigned* phase) {
  __syncthreads();
  if (threadIdx.x == 0) {
    __threadfence();
    __hip_atomic_fetch_add(ctr, 1u, __ATOMIC_RELEASE, __HIP_MEMORY_SCOPE_AGENT);
    unsigned tgt = (++(*phase)) * (unsigned)NB_SCAN;
    while (__hip_atomic_load(ctr, __ATOMIC_ACQUIRE, __HIP_MEMORY_SCOPE_AGENT) < tgt)
      __builtin_amdgcn_s_sleep(1);
  } else {
    ++(*phase);
  }
  __syncthreads();
}

// Block nb owns cols [nb*64, +64) of each gate; wave (m=w%4, sel=w/4) owns 32
// of them -> 6 accumulators; gates fully register-resident. Register
// double-buffered fragment pipeline; one grid barrier per timestep.
__global__ void __launch_bounds__(256) k_scan(
    const float* __restrict__ xg, const unsigned short* __restrict__ whh_sw,
    const float* __restrict__ bhh, unsigned short* __restrict__ h_bf,
    float* __restrict__ h_f32, unsigned short* __restrict__ y_out,
    unsigned* __restrict__ ctr) {
  const int tid = threadIdx.x;
  const int nb = blockIdx.x;
  const int w = tid >> 5, lane = tid & 31, lane16 = lane & 15, half = lane >> 4;
  const int m16 = (w & 3) * 16, sel = w >> 2;

  for (int i = blockIdx.x * 256 + tid; i < B_ * H; i += NB_SCAN * 256) h_bf[i] = 0;
  unsigned phase = 0;
  grid_bar(ctr, &phase);

  float bias[3][2];
#pragma unroll
  for (int g = 0; g < 3; ++g)
#pragma unroll
    for (int j = 0; j < 2; ++j)
      bias[g][j] = bhh[g * H + nb * 64 + sel * 32 + j * 16 + lane16];

  float hreg[2][8];
#pragma unroll
  for (int j = 0; j < 2; ++j)
#pragma unroll
    for (int r = 0; r < 8; ++r) hreg[j][r] = 0.0f;

  const unsigned short* arow = h_bf + (size_t)(m16 + lane16) * H + half * 8;
  const unsigned short* bbase = whh_sw + ((size_t)(nb * 4 + sel * 2) * 32 + lane) * 16;
  const size_t BSTRIDE = (size_t)NT_W * 512;               // elements per kt step
  const int boff[6] = {0 * 512, 1 * 512, 64 * 512, 65 * 512, 128 * 512, 129 * 512};

  for (int t = 0; t < T_; ++t) {
    v8f acc[3][2] = {};
    const unsigned short* ap = arow;
    const unsigned short* bp = bbase;
    v16bf a_cur = ld_afrag(ap);
    v16bf b_cur[6];
#pragma unroll
    for (int q = 0; q < 6; ++q) b_cur[q] = *(const v16bf*)(bp + boff[q]);

    for (int kt = 0; kt < H / 32; ++kt) {
      v16bf a_nxt = a_cur;
      v16bf b_nxt[6];
      if (kt + 1 < H / 32) {          // fetch kt+1 while issuing kt's WMMAs
        ap += 32;
        bp += BSTRIDE;
        a_nxt = ld_afrag(ap);
#pragma unroll
        for (int q = 0; q < 6; ++q) b_nxt[q] = *(const v16bf*)(bp + boff[q]);
      } else {
#pragma unroll
        for (int q = 0; q < 6; ++q) b_nxt[q] = b_cur[q];
      }
#pragma unroll
      for (int g = 0; g < 3; ++g)
#pragma unroll
        for (int j = 0; j < 2; ++j)
          acc[g][j] = __builtin_amdgcn_wmma_f32_16x16x32_bf16(
              false, a_cur, false, b_cur[g * 2 + j], (short)0, acc[g][j], false, false);
      a_cur = a_nxt;
#pragma unroll
      for (int q = 0; q < 6; ++q) b_cur[q] = b_nxt[q];
    }

#pragma unroll
    for (int j = 0; j < 2; ++j) {
      const int c = nb * 64 + sel * 32 + j * 16 + lane16;
      const size_t xb0 = (size_t)(m16 + half * 8) * T_ * G + (size_t)t * G + c;
      if (t + 1 < T_) __builtin_prefetch(&xg[xb0 + G], 0, 1);
#pragma unroll
      for (int r = 0; r < 8; ++r) {
        const int brow = m16 + r + half * 8;
        const size_t xb = xb0 + (size_t)r * T_ * G;
        float xr = xg[xb], xz = xg[xb + H], xn = xg[xb + 2 * H];
        float rr = fsigmoid(xr + acc[0][j][r] + bias[0][j]);
        float zz = fsigmoid(xz + acc[1][j][r] + bias[1][j]);
        float nn = ftanh_(xn + rr * (acc[2][j][r] + bias[2][j]));
        float hn = (1.0f - zz) * nn + zz * hreg[j][r];
        hreg[j][r] = hn;
        unsigned short hb = f2bf(hn);
        h_bf[brow * H + c] = hb;
        if (y_out) y_out[(size_t)(brow * T_ + t) * H + c] = hb;
      }
    }
    grid_bar(ctr, &phase);
  }
#pragma unroll
  for (int j = 0; j < 2; ++j) {
    const int c = nb * 64 + sel * 32 + j * 16 + lane16;
#pragma unroll
    for (int r = 0; r < 8; ++r) h_f32[(m16 + r + half * 8) * H + c] = hreg[j][r];
  }
}

__global__ void k_final(const float* __restrict__ h, const float* __restrict__ Wl,
                        const float* __restrict__ bl, float* __restrict__ out) {
  int b = threadIdx.x;
  if (b < B_) {
    float s = 0.0f;
    for (int k = 0; k < H; ++k) s += h[b * H + k] * Wl[k];
    out[b] = s + bl[0];
  }
}

extern "C" void kernel_launch(void* const* d_in, const int* in_sizes, int n_in,
                              void* d_out, int out_size, void* d_ws, size_t ws_size,
                              hipStream_t stream) {
  const float* x     = (const float*)d_in[0];
  const float* Wih0  = (const float*)d_in[1];
  const float* Wih12 = (const float*)d_in[2];
  const float* Whh   = (const float*)d_in[3];
  const float* bih   = (const float*)d_in[4];
  const float* bhh   = (const float*)d_in[5];
  const float* Wlin  = (const float*)d_in[6];
  const float* blin  = (const float*)d_in[7];
  float* out = (float*)d_out;

  char* ws = (char*)d_ws;
  size_t cur = 0;
  auto carve = [&](size_t bytes) -> void* {
    void* p = ws + cur;
    cur += (bytes + 255) & ~(size_t)255;
    return p;
  };
  float* xg           = (float*)carve((size_t)BT * G * 4);
  unsigned short* xbf = (unsigned short*)carve((size_t)BT * 128 * 2);
  unsigned short* y0  = (unsigned short*)carve((size_t)BT * H * 2);
  unsigned short* y1  = (unsigned short*)carve((size_t)BT * H * 2);
  unsigned short* wih_sw = (unsigned short*)carve((size_t)32 * NT_W * 1024);
  unsigned short* whh_sw = (unsigned short*)carve((size_t)32 * NT_W * 1024);
  unsigned short* h_bf   = (unsigned short*)carve((size_t)B_ * H * 2);
  float* h_f32           = (float*)carve((size_t)B_ * H * 4);
  unsigned* ctr          = (unsigned*)carve(256);
  if (cur > ws_size) return;

  const dim3 blk(256);
  const int swz_full = (32 * NT_W * 32 + 255) / 256;

  k_convert_x<<<(BT * 128 + 255) / 256, blk, 0, stream>>>(x, xbf);

  // ---- Layer 0 (Din = 118 padded to 128) ----
  k_swizzle_w<<<(4 * NT_W * 32 + 255) / 256, blk, 0, stream>>>(Wih0, 118, 4, (unsigned*)wih_sw);
  k_swizzle_w<<<swz_full, blk, 0, stream>>>(Whh, 1024, 32, (unsigned*)whh_sw);
  k_input_gemm<<<dim3(BT / 64, G / 128), blk, 0, stream>>>(xbf, 128, wih_sw, bih, xg);
  (void)hipMemsetAsync(ctr, 0, 16, stream);
  k_scan<<<NB_SCAN, blk, 0, stream>>>(xg, whh_sw, bhh, h_bf, h_f32, y0, ctr);

  // ---- Layer 1 ----
  k_swizzle_w<<<swz_full, blk, 0, stream>>>(Wih12, 1024, 32, (unsigned*)wih_sw);
  k_swizzle_w<<<swz_full, blk, 0, stream>>>(Whh + (size_t)G * H, 1024, 32, (unsigned*)whh_sw);
  k_input_gemm<<<dim3(BT / 64, G / 128), blk, 0, stream>>>(y0, 1024, wih_sw, bih + G, xg);
  (void)hipMemsetAsync(ctr, 0, 16, stream);
  k_scan<<<NB_SCAN, blk, 0, stream>>>(xg, whh_sw, bhh + G, h_bf, h_f32, y1, ctr);

  // ---- Layer 2 ----
  k_swizzle_w<<<swz_full, blk, 0, stream>>>(Wih12 + (size_t)G * H, 1024, 32, (unsigned*)wih_sw);
  k_swizzle_w<<<swz_full, blk, 0, stream>>>(Whh + 2 * (size_t)G * H, 1024, 32, (unsigned*)whh_sw);
  k_input_gemm<<<dim3(BT / 64, G / 128), blk, 0, stream>>>(y1, 1024, wih_sw, bih + 2 * G, xg);
  (void)hipMemsetAsync(ctr, 0, 16, stream);
  k_scan<<<NB_SCAN, blk, 0, stream>>>(xg, whh_sw, bhh + 2 * G, h_bf, h_f32, nullptr, ctr);

  k_final<<<1, 64, 0, stream>>>(h_f32, Wlin, blin, out);
}
